// gat_net_28363964022953
// MI455X (gfx1250) — compile-verified, matrix operand order checked by reading, not run on hardware
//
#include <hip/hip_runtime.h>
#include <hip/hip_bf16.h>

typedef __attribute__((ext_vector_type(16))) _Float16 v16h;
typedef __attribute__((ext_vector_type(8)))  float    v8f;

#define CDIM 128

// ---------------------------------------------------------------------------
// Kernel 1: fuse  Wf = W_type @ gat_W  (and bf = b_type @ gat_W), writing Wf
// directly in CDNA5 WMMA B-fragment layout (f16): frag[kstep][tile][lane][16].
// Per ISA 16-bit operand layout: lane L holds column n = L%16; half-group
// hk = L/16 holds k-offsets {0..7,16..23} (+8 for hk=1) within a 32-k step.
// ---------------------------------------------------------------------------
__global__ void fuse_swizzle_kernel(const float* __restrict__ W,    // [K,128]
                                    const float* __restrict__ bias, // [128]
                                    const float* __restrict__ gatW, // [128,128]
                                    int K, int Kpad,
                                    _Float16* __restrict__ frag,    // [Kpad/32][8][32][16]
                                    float* __restrict__ bfused)     // [128]
{
    int idx = blockIdx.x * blockDim.x + threadIdx.x;
    int total = (Kpad + 1) * CDIM;
    if (idx >= total) return;
    int k = idx / CDIM;
    int c = idx % CDIM;
    if (k == Kpad) {
        float v = 0.f;
        for (int j = 0; j < CDIM; ++j) v += bias[j] * gatW[j * CDIM + c];
        bfused[c] = v;
        return;
    }
    float v = 0.f;
    if (k < K) {
        for (int j = 0; j < CDIM; ++j) v += W[k * CDIM + j] * gatW[j * CDIM + c];
    }
    int s    = k >> 5;
    int kis  = k & 31;
    int hk   = (kis >> 3) & 1;            // 8..15 / 24..31 -> upper half-group
    int k0   = kis - 8 * hk;              // {0..7, 16..23}
    int i    = (k0 < 8) ? (k0 >> 1) : (4 + ((k0 - 16) >> 1));
    int elem = 2 * i + (k0 & 1);
    int lane = hk * 16 + (c & 15);
    int t    = c >> 4;
    frag[(((size_t)(s * 8 + t)) * 32 + lane) * 16 + elem] = (_Float16)v;
}

// ---------------------------------------------------------------------------
// Kernel 2: one wave computes a 16-row x 128-col strip of
//   x[node] = emb_row @ Wf + bf   (scattered to x[b, set[row], :]),
// and fuses alpha_src = x . att_src, alpha_dst = x . att_dst in-register.
// Full k-steps use unconditional loads; only the tail step is guarded.
// ---------------------------------------------------------------------------
__global__ __launch_bounds__(32) void xcalc_kernel(
    const float*    __restrict__ emb,    // [B, M, K]
    const int*      __restrict__ setids, // [B, M]
    const _Float16* __restrict__ frag,   // swizzled fused weights
    const float*    __restrict__ bfused, // [128]
    const float*    __restrict__ att_src,
    const float*    __restrict__ att_dst,
    int M, int K, int N1,
    float* __restrict__ x_out,           // [B, N1, 128]
    float* __restrict__ asrc,            // [B, N1]
    float* __restrict__ adst)            // [B, N1]
{
    int b     = blockIdx.y;
    int strip = blockIdx.x;
    int lane  = threadIdx.x;
    int n16   = lane & 15;
    int hk    = lane >> 4;

    const float* arow = emb + ((size_t)b * M + (size_t)strip * 16 + n16) * K;

    float atts[8], attd[8], bfl[8];
#pragma unroll
    for (int t = 0; t < 8; ++t) {
        atts[t] = att_src[t * 16 + n16];
        attd[t] = att_dst[t * 16 + n16];
        bfl[t]  = bfused[t * 16 + n16];
    }

    v8f acc[8] = {};

    int ksfull = K >> 5;
    for (int s = 0; s < ksfull; ++s) {
        // A fragment: elements 0..7 = k kb..kb+7, elements 8..15 = kb+16..kb+23
        const float* ap = arow + s * 32 + hk * 8;
        v16h af;
#pragma unroll
        for (int i = 0; i < 8; ++i) {
            int k = (i < 4) ? (2 * i) : (16 + 2 * (i - 4));
            af[2 * i]     = (_Float16)ap[k];
            af[2 * i + 1] = (_Float16)ap[k + 1];
        }
        // B fragments: 8 tiles, contiguous 32B per lane (pre-swizzled) --
        // load all first so the clause overlaps with the WMMA drain.
        const _Float16* fb = frag + ((size_t)s * 256 + lane) * 16;
        v16h bfr[8];
#pragma unroll
        for (int t = 0; t < 8; ++t) bfr[t] = *(const v16h*)(fb + (size_t)t * 512);
#pragma unroll
        for (int t = 0; t < 8; ++t)
            acc[t] = __builtin_amdgcn_wmma_f32_16x16x32_f16(
                false, af, false, bfr[t], (short)0, acc[t], false, false);
    }
    if (K & 31) {                         // guarded tail step
        int s = ksfull;
        int kb = s * 32 + hk * 8;
        v16h af;
#pragma unroll
        for (int i = 0; i < 8; ++i) {
            int k = kb + ((i < 4) ? (2 * i) : (16 + 2 * (i - 4)));
            float x0 = (k     < K) ? arow[k]     : 0.f;
            float x1 = (k + 1 < K) ? arow[k + 1] : 0.f;
            af[2 * i]     = (_Float16)x0;
            af[2 * i + 1] = (_Float16)x1;
        }
        const _Float16* fb = frag + ((size_t)s * 256 + lane) * 16;
        v16h bfr[8];
#pragma unroll
        for (int t = 0; t < 8; ++t) bfr[t] = *(const v16h*)(fb + (size_t)t * 512);
#pragma unroll
        for (int t = 0; t < 8; ++t)
            acc[t] = __builtin_amdgcn_wmma_f32_16x16x32_f16(
                false, af, false, bfr[t], (short)0, acc[t], false, false);
    }

    // D layout: acc[t][r] = x[row = r + 8*hk, col = t*16 + n16]
    float ps[8], pd[8];
#pragma unroll
    for (int r = 0; r < 8; ++r) { ps[r] = 0.f; pd[r] = 0.f; }
#pragma unroll
    for (int t = 0; t < 8; ++t) {
#pragma unroll
        for (int r = 0; r < 8; ++r) {
            float xv = acc[t][r] + bfl[t];
            acc[t][r] = xv;
            ps[r] += xv * atts[t];
            pd[r] += xv * attd[t];
        }
    }
    // reduce alpha partials across the 16 column lanes (stays in half-group)
#pragma unroll
    for (int r = 0; r < 8; ++r) {
#pragma unroll
        for (int off = 1; off < 16; off <<= 1) {
            ps[r] += __shfl_xor(ps[r], off, 32);
            pd[r] += __shfl_xor(pd[r], off, 32);
        }
    }

    int rowbase = strip * 16 + hk * 8;
#pragma unroll
    for (int r = 0; r < 8; ++r) {
        int node = setids[(size_t)b * M + rowbase + r];
        size_t xb = ((size_t)b * N1 + node) * CDIM;
#pragma unroll
        for (int t = 0; t < 8; ++t) x_out[xb + t * 16 + n16] = acc[t][r];
        if (n16 == 0) {
            asrc[(size_t)b * N1 + node] = ps[r];
            adst[(size_t)b * N1 + node] = pd[r];
        }
    }
}

// ---------------------------------------------------------------------------
// Edge pass 1: e = leaky_relu(a_src[src] + a_dst[dst]); segment-max via
// order-preserving float->u32 encoding + atomicMax.
// ---------------------------------------------------------------------------
__global__ void edge_pass1_kernel(const int* __restrict__ ei,
                                  const float* __restrict__ asrc,
                                  const float* __restrict__ adst,
                                  float* __restrict__ ebuf,
                                  unsigned* __restrict__ emax,
                                  int E, int EE, int N1, int B)
{
    size_t idx = (size_t)blockIdx.x * blockDim.x + threadIdx.x;
    if (idx >= (size_t)B * EE) return;
    int b = (int)(idx / EE);
    int i = (int)(idx % EE);
    int s, d;
    if (i < E) {
        s = ei[(size_t)b * 2 * E + i];
        d = ei[(size_t)b * 2 * E + E + i];
    } else {
        s = d = i - E;                    // self loops (node 0..N1-1)
    }
    float e = asrc[(size_t)b * N1 + s] + adst[(size_t)b * N1 + d];
    e = (e > 0.f) ? e : 0.2f * e;
    ebuf[idx] = e;
    unsigned bits = __float_as_uint(e);
    unsigned enc = (bits & 0x80000000u) ? ~bits : (bits | 0x80000000u);
    atomicMax(&emax[(size_t)b * N1 + d], enc);
}

// ---------------------------------------------------------------------------
// Edge pass 2: wave per edge. ee = exp(e - max); scatter ee*x[src] into
// accum[dst] (4 channels/lane) and ee into denom[dst].
// ---------------------------------------------------------------------------
__global__ __launch_bounds__(256) void edge_pass2_kernel(
    const int* __restrict__ ei, const float* __restrict__ ebuf,
    const unsigned* __restrict__ emax,
    const float* __restrict__ x, float* __restrict__ accum,
    float* __restrict__ denom, int E, int EE, int N1, int B)
{
    size_t wid = ((size_t)blockIdx.x * blockDim.x + threadIdx.x) >> 5;
    int lane = threadIdx.x & 31;
    if (wid >= (size_t)B * EE) return;
    int b = (int)(wid / EE);
    int i = (int)(wid % EE);
    int s, d;
    if (i < E) {
        s = ei[(size_t)b * 2 * E + i];
        d = ei[(size_t)b * 2 * E + E + i];
    } else {
        s = d = i - E;
    }
    float e = ebuf[wid];
    unsigned enc = emax[(size_t)b * N1 + d];
    unsigned mbits = (enc & 0x80000000u) ? (enc & 0x7FFFFFFFu) : ~enc;
    float ee = __expf(e - __uint_as_float(mbits));
    const float* xs = x + ((size_t)b * N1 + s) * CDIM;
    float* ad = accum + ((size_t)b * N1 + d) * CDIM;
    int c0 = lane * 4;
    float4 xv = *(const float4*)(xs + c0);
    atomicAdd(ad + c0 + 0, ee * xv.x);
    atomicAdd(ad + c0 + 1, ee * xv.y);
    atomicAdd(ad + c0 + 2, ee * xv.z);
    atomicAdd(ad + c0 + 3, ee * xv.w);
    if (lane == 0) atomicAdd(denom + (size_t)b * N1 + d, ee);
}

// ---------------------------------------------------------------------------
// h = relu(accum/denom + gat_b); h[0] = 1e-5 (in place over accum)
// ---------------------------------------------------------------------------
__global__ void finish_h_kernel(float* __restrict__ accum,
                                const float* __restrict__ denom,
                                const float* __restrict__ gat_b,
                                int N1, size_t total)
{
    size_t idx = (size_t)blockIdx.x * blockDim.x + threadIdx.x;
    if (idx >= total) return;
    size_t n = idx >> 7;
    int c = (int)(idx & 127);
    float v = accum[idx] / (denom[n] + 1e-16f) + gat_b[c];
    v = (v > 0.f) ? v : 0.f;
    if ((n % (size_t)N1) == 0) v = 1e-5f;
    accum[idx] = v;
}

// ---------------------------------------------------------------------------
// Pooling + 3-layer MLP, one block per graph.
// cat = [h[aid0] | sum(h[aid1:])/(acnt-1.00001) | sum(h[tid])/(tcnt+1e-5)]
// ---------------------------------------------------------------------------
__global__ __launch_bounds__(256) void pool_mlp_kernel(
    const float* __restrict__ h,          // [B, N1, 128]
    const int* __restrict__ tid_arr, const int* __restrict__ aid_arr,
    const float* __restrict__ tcnt, const float* __restrict__ acnt,
    const float* __restrict__ W1, const float* __restrict__ b1,
    const float* __restrict__ W2, const float* __restrict__ b2,
    const float* __restrict__ W3, const float* __restrict__ b3,
    int N1, int n_tid, int n_aid, float* __restrict__ out)
{
    __shared__ float cat[3 * CDIM];
    __shared__ float h1s[2 * CDIM];
    __shared__ float h2s[CDIM];
    int b = blockIdx.x;
    int tid = threadIdx.x;
    const float* hb = h + (size_t)b * N1 * CDIM;
    if (tid < CDIM) {
        int c = tid;
        float tp = 0.f;
        for (int j = 0; j < n_tid; ++j)
            tp += hb[(size_t)tid_arr[b * n_tid + j] * CDIM + c];
        cat[2 * CDIM + c] = tp / (1e-5f + tcnt[b]);
        cat[c] = hb[(size_t)aid_arr[b * n_aid] * CDIM + c];
        float as = 0.f;
        for (int j = 1; j < n_aid; ++j)
            as += hb[(size_t)aid_arr[b * n_aid + j] * CDIM + c];
        cat[CDIM + c] = as / (-1.00001f + acnt[b]);
    }
    __syncthreads();
    {
        int o = tid;                       // 0..255
        float v = b1[o];
        for (int k = 0; k < 3 * CDIM; ++k) v += cat[k] * W1[k * 2 * CDIM + o];
        h1s[o] = tanhf(v);
    }
    __syncthreads();
    if (tid < CDIM) {
        float v = b2[tid];
        for (int k = 0; k < 2 * CDIM; ++k) v += h1s[k] * W2[k * CDIM + tid];
        h2s[tid] = tanhf(v);
    }
    __syncthreads();
    if (tid == 0) {
        float v = b3[0];
        for (int k = 0; k < CDIM; ++k) v += h2s[k] * W3[k];
        out[b] = v;
    }
}

// ---------------------------------------------------------------------------
extern "C" void kernel_launch(void* const* d_in, const int* in_sizes, int n_in,
                              void* d_out, int out_size, void* d_ws, size_t ws_size,
                              hipStream_t stream)
{
    (void)n_in; (void)out_size; (void)ws_size;
    const float* topic_emb = (const float*)d_in[0];
    const float* au_emb    = (const float*)d_in[1];
    const float* paper_emb = (const float*)d_in[2];
    const int* topic_set   = (const int*)d_in[3];
    const int* author_set  = (const int*)d_in[4];
    const int* paper_set   = (const int*)d_in[5];
    const int* edge_index  = (const int*)d_in[6];
    const int* topic_id    = (const int*)d_in[7];
    const int* author_id   = (const int*)d_in[8];
    const float* topic_cnt = (const float*)d_in[9];
    const float* auth_cnt  = (const float*)d_in[10];
    const float* W_topic = (const float*)d_in[11];
    const float* b_topic = (const float*)d_in[12];
    const float* W_au    = (const float*)d_in[13];
    const float* b_au    = (const float*)d_in[14];
    const float* W_paper = (const float*)d_in[15];
    const float* b_paper = (const float*)d_in[16];
    const float* gat_W   = (const float*)d_in[17];
    const float* att_src = (const float*)d_in[18];
    const float* att_dst = (const float*)d_in[19];
    const float* gat_b   = (const float*)d_in[20];
    const float* W1 = (const float*)d_in[21];
    const float* b1 = (const float*)d_in[22];
    const float* W2 = (const float*)d_in[23];
    const float* b2 = (const float*)d_in[24];
    const float* W3 = (const float*)d_in[25];
    const float* b3 = (const float*)d_in[26];

    int B  = in_sizes[9];
    int T  = in_sizes[3] / B;
    int A  = in_sizes[4] / B;
    int P  = in_sizes[5] / B;
    int E  = in_sizes[6] / (2 * B);
    int N1 = T + A + P + 1;
    int KT = in_sizes[11] / CDIM;
    int KA = in_sizes[13] / CDIM;
    int KP = in_sizes[15] / CDIM;
    int n_tid = in_sizes[7] / B;
    int n_aid = in_sizes[8] / B;
    int EE = E + N1;
    int kstT = (KT + 31) / 32, kstA = (KA + 31) / 32, kstP = (KP + 31) / 32;

    // --- carve workspace ---
    char* ws = (char*)d_ws;
    size_t off = 0;
    auto alloc = [&](size_t bytes) -> void* {
        void* p = ws + off;
        off = (off + bytes + 255) & ~(size_t)255;
        return p;
    };
    _Float16* fragT = (_Float16*)alloc((size_t)kstT * 8 * 32 * 16 * 2);
    _Float16* fragA = (_Float16*)alloc((size_t)kstA * 8 * 32 * 16 * 2);
    _Float16* fragP = (_Float16*)alloc((size_t)kstP * 8 * 32 * 16 * 2);
    float* bfT = (float*)alloc(CDIM * 4);
    float* bfA = (float*)alloc(CDIM * 4);
    float* bfP = (float*)alloc(CDIM * 4);
    float*    xbuf  = (float*)alloc((size_t)B * N1 * CDIM * 4);
    float*    asrc  = (float*)alloc((size_t)B * N1 * 4);
    float*    adst  = (float*)alloc((size_t)B * N1 * 4);
    float*    ebuf  = (float*)alloc((size_t)B * EE * 4);
    unsigned* emax  = (unsigned*)alloc((size_t)B * N1 * 4);
    float*    denom = (float*)alloc((size_t)B * N1 * 4);
    float*    accum = (float*)alloc((size_t)B * N1 * CDIM * 4);

    hipMemsetAsync(xbuf,  0, (size_t)B * N1 * CDIM * 4, stream); // x[0] must be 0
    hipMemsetAsync(asrc,  0, (size_t)B * N1 * 4, stream);
    hipMemsetAsync(adst,  0, (size_t)B * N1 * 4, stream);
    hipMemsetAsync(emax,  0, (size_t)B * N1 * 4, stream);
    hipMemsetAsync(denom, 0, (size_t)B * N1 * 4, stream);
    hipMemsetAsync(accum, 0, (size_t)B * N1 * CDIM * 4, stream);

    // 1) fuse + swizzle weights
    auto launch_fuse = [&](const float* W, const float* bias, int K, int kst,
                           _Float16* frag, float* bf) {
        int total = (kst * 32 + 1) * CDIM;
        fuse_swizzle_kernel<<<(total + 255) / 256, 256, 0, stream>>>(
            W, bias, gat_W, K, kst * 32, frag, bf);
    };
    launch_fuse(W_topic, b_topic, KT, kstT, fragT, bfT);
    launch_fuse(W_au,    b_au,    KA, kstA, fragA, bfA);
    launch_fuse(W_paper, b_paper, KP, kstP, fragP, bfP);

    // 2) WMMA GEMMs: x rows + fused alpha projections, scattered by node id
    auto launch_x = [&](const float* emb, const int* setids, int M, int K,
                        const _Float16* frag, const float* bf) {
        dim3 grid(M / 16, B);
        xcalc_kernel<<<grid, 32, 0, stream>>>(emb, setids, frag, bf, att_src, att_dst,
                                              M, K, N1, xbuf, asrc, adst);
    };
    launch_x(topic_emb, topic_set,  T, KT, fragT, bfT);
    launch_x(au_emb,    author_set, A, KA, fragA, bfA);
    launch_x(paper_emb, paper_set,  P, KP, fragP, bfP);

    // 3) edge softmax aggregation
    size_t nedge = (size_t)B * EE;
    edge_pass1_kernel<<<(unsigned)((nedge + 255) / 256), 256, 0, stream>>>(
        edge_index, asrc, adst, ebuf, emax, E, EE, N1, B);
    edge_pass2_kernel<<<(unsigned)((nedge * 32 + 255) / 256), 256, 0, stream>>>(
        edge_index, ebuf, emax, xbuf, accum, denom, E, EE, N1, B);
    size_t total_h = (size_t)B * N1 * CDIM;
    finish_h_kernel<<<(unsigned)((total_h + 255) / 256), 256, 0, stream>>>(
        accum, denom, gat_b, N1, total_h);

    // 4) pooling + MLP
    pool_mlp_kernel<<<B, 256, 0, stream>>>(accum, topic_id, author_id, topic_cnt,
                                           auth_cnt, W1, b1, W2, b2, W3, b3,
                                           N1, n_tid, n_aid, (float*)d_out);
}
